// RNNcoder_69406671504086
// MI455X (gfx1250) — compile-verified, matrix operand order checked by reading, not run on hardware
//
#include <hip/hip_runtime.h>
#include <hip/hip_bf16.h>

// GRU-style encoder for MI455X (gfx1250, wave32, WMMA, async-LDS).
// B=64, T=200, V=32000, U=1024.
//
// Pipeline (all on `stream`, dependencies via stream order):
//   1) transpose_bf16_kernel x3 : Ww/Wu/Wh fp32 [K][N] -> bf16 [N][K] in ws
//   2) embed_proj_kernel        : pre[t][b][:] = E[X[b,t]] @ Ww + bw
//                                 (bf16 WMMA, async global->LDS double buffer)
//   3) rnn_scan_kernel          : 200-step scan; batch rows partitioned across
//                                 4 workgroups (recurrence independent per row)
//
// Workspace layout:
//   [0,2MB)   Wwt bf16   [2MB,4MB) Wut bf16   [4MB,6MB) Wht bf16
//   [6MB, 6MB+52MB)  pre fp32 [T][B][U]

#define Bsz 64
#define Tsz 200
#define Usz 1024

#define GLOBAL_AS __attribute__((address_space(1)))
#define LDS_AS    __attribute__((address_space(3)))

#if __has_builtin(__builtin_amdgcn_global_load_async_to_lds_b128)
#define HAVE_ASYNC_LDS 1
#else
#define HAVE_ASYNC_LDS 0
#endif

typedef __attribute__((ext_vector_type(4)))  int    int4v;
typedef __attribute__((ext_vector_type(16))) __bf16 bf16x16;
typedef __attribute__((ext_vector_type(8)))  float  floatx8;

union FragBF {
  bf16x16 v;
  uint4 q[2];
  unsigned ui[8];
  unsigned short us[16];
};

__device__ __forceinline__ unsigned short f2bf(float f) {
  union { float f; unsigned u; } v; v.f = f;
  unsigned r = v.u + 0x7fffu + ((v.u >> 16) & 1u);   // round-to-nearest-even
  return (unsigned short)(r >> 16);
}

__device__ __forceinline__ unsigned pack_bf2(float lo, float hi) {
  return (unsigned)f2bf(lo) | ((unsigned)f2bf(hi) << 16);
}

__device__ __forceinline__ floatx8 wmma_bf16(const FragBF& a, const FragBF& b, floatx8 c) {
  return __builtin_amdgcn_wmma_f32_16x16x32_bf16(
      /*neg_a=*/false, a.v, /*neg_b=*/false, b.v,
      /*c_mod=*/(short)0, c, /*reuse_a=*/false, /*reuse_b=*/false);
}

// Wait until this wave's outstanding async-LDS copies are <= N.
template <int N>
__device__ __forceinline__ void wait_async_le() {
#if HAVE_ASYNC_LDS
#if __has_builtin(__builtin_amdgcn_s_wait_asynccnt)
  __builtin_amdgcn_s_wait_asynccnt((unsigned short)N);
#else
  if (N == 0) asm volatile("s_wait_asynccnt 0x0" ::: "memory");
  else        asm volatile("s_wait_asynccnt 0x4" ::: "memory");
#endif
#endif
}

// Copy 64 contiguous bytes global -> LDS (async if available).
__device__ __forceinline__ void stage64(const float* __restrict__ g, float* l) {
#if HAVE_ASYNC_LDS
  GLOBAL_AS int4v* gp = (GLOBAL_AS int4v*)g;
  LDS_AS    int4v* lp = (LDS_AS int4v*)l;
  __builtin_amdgcn_global_load_async_to_lds_b128(gp, lp, 0,  0);
  __builtin_amdgcn_global_load_async_to_lds_b128(gp, lp, 16, 0);
  __builtin_amdgcn_global_load_async_to_lds_b128(gp, lp, 32, 0);
  __builtin_amdgcn_global_load_async_to_lds_b128(gp, lp, 48, 0);
#else
  const float4* s = (const float4*)g;
  float4* d = (float4*)l;
  d[0] = s[0]; d[1] = s[1]; d[2] = s[2]; d[3] = s[3];
#endif
}

// Build a bf16 A fragment from an fp32 LDS row (CDNA5 16-bit A layout:
// elements 0..7 -> K = hi*8..hi*8+7, elements 8..15 -> K = 16+hi*8..).
__device__ __forceinline__ FragBF frag_from_f32row(const float* arow, int hi) {
  const float4* p0 = (const float4*)(arow + hi * 8);
  const float4* p1 = (const float4*)(arow + 16 + hi * 8);
  float4 a0 = p0[0], a1 = p0[1], a2 = p1[0], a3 = p1[1];
  FragBF a;
  a.ui[0] = pack_bf2(a0.x, a0.y); a.ui[1] = pack_bf2(a0.z, a0.w);
  a.ui[2] = pack_bf2(a1.x, a1.y); a.ui[3] = pack_bf2(a1.z, a1.w);
  a.ui[4] = pack_bf2(a2.x, a2.y); a.ui[5] = pack_bf2(a2.z, a2.w);
  a.ui[6] = pack_bf2(a3.x, a3.y); a.ui[7] = pack_bf2(a3.z, a3.w);
  return a;
}

// ---------------------------------------------------------------------------
// 1) fp32 [K][N] -> bf16 [N][K] (writes coalesced; reads served by L2)
// ---------------------------------------------------------------------------
__global__ __launch_bounds__(256)
void transpose_bf16_kernel(const float* __restrict__ src,
                           unsigned short* __restrict__ dst) {
  int i = blockIdx.x * 256 + threadIdx.x;      // i over U*U
  int n = i >> 10;
  int k = i & (Usz - 1);
  dst[i] = f2bf(src[k * Usz + n]);
}

// ---------------------------------------------------------------------------
// 2) pre = gather(E) @ Ww + bw.  Block: 256 thr (8 waves), tile 128(M) x 64(N).
//    A tiles (128x32 fp32) streamed global->LDS with ASYNC double buffering;
//    B fragments straight from L2-resident transposed bf16 weights.
// ---------------------------------------------------------------------------
__global__ __launch_bounds__(256)
void embed_proj_kernel(const int* __restrict__ X, const float* __restrict__ E,
                       const unsigned short* __restrict__ Wwt,
                       const float* __restrict__ bw, float* __restrict__ pre) {
  __shared__ alignas(16) float Af[2][128 * 32];
  __shared__ int Xrow[128];

  const int tid = threadIdx.x;
  const int m0  = blockIdx.x * 128;           // flat m = t*B + b
  const int n0b = blockIdx.y * 64;

  if (tid < 128) {
    int m = m0 + tid;
    int t = m >> 6;                            // B == 64
    int b = m & 63;
    Xrow[tid] = X[b * Tsz + t];
  }
  __syncthreads();

  const int lane = tid & 31, wave = tid >> 5;
  const int hi = lane >> 4, nl = lane & 15;
  const int mrow = wave * 16;

  // staging role: 256 threads cover 128 rows x two 16-float halves
  const int srow = tid >> 1, shalf = tid & 1;
  const float* esrc = E + (long)Xrow[srow] * Usz + shalf * 16;
  float* const sdst0 = &Af[0][srow * 32 + shalf * 16];
  float* const sdst1 = &Af[1][srow * 32 + shalf * 16];

  floatx8 acc[4] = {};

  stage64(esrc, sdst0);                        // prologue: stage kb = 0
  int cur = 0;

  for (int kb = 0; kb < Usz; kb += 32) {
    if (kb + 32 < Usz) {                       // stage next tile into other buf
      stage64(esrc + kb + 32, cur ? sdst0 : sdst1);
      wait_async_le<4>();                      // current tile's 4 copies done
    } else {
      wait_async_le<0>();
    }
    __syncthreads();

    const float* arow = &Af[cur][(mrow + nl) * 32];
    FragBF a = frag_from_f32row(arow, hi);

#pragma unroll
    for (int tn = 0; tn < 4; tn++) {
      const int n = n0b + tn * 16 + nl;
      const uint4* bp = (const uint4*)(Wwt + n * Usz + kb);
      FragBF bm;
      bm.q[0] = bp[hi];
      bm.q[1] = bp[2 + hi];
      acc[tn] = wmma_bf16(a, bm, acc[tn]);
    }
    __syncthreads();
    cur ^= 1;
  }

#pragma unroll
  for (int tn = 0; tn < 4; tn++) {
    const int n = n0b + tn * 16 + nl;
    const float bwv = bw[n];
#pragma unroll
    for (int r = 0; r < 8; r++) {
      const int m = m0 + mrow + r + 8 * hi;
      pre[(long)m * Usz + n] = acc[tn][r] + bwv;   // re-read once by the scan
    }
  }
}

// ---------------------------------------------------------------------------
// 3) Persistent recurrence.  4 blocks x 512 thr (16 waves).  Block owns batch
//    rows gb0..gb0+15; wave owns 64 output columns (4 WMMA N-tiles).  h lives
//    in accumulator-layout registers; bf16 mirrors of h and z*h in LDS feed
//    the A operand of both per-step GEMMs.  Weights stay L2-resident (2 MB
//    each).  pre is read non-temporally; hs is stored non-temporally so the
//    192MB L2 keeps the recurrence working set.
// ---------------------------------------------------------------------------
__global__ __launch_bounds__(512)
void rnn_scan_kernel(const int* __restrict__ X, const float* __restrict__ E,
                     const float* __restrict__ hidden,
                     const unsigned short* __restrict__ Wut,
                     const unsigned short* __restrict__ Wht,
                     const float* __restrict__ bu, const float* __restrict__ bc,
                     const float* __restrict__ bh, const float* __restrict__ pre,
                     float* __restrict__ out) {
  __shared__ alignas(16) unsigned short hbf[16 * Usz];   // h   chunk, bf16
  __shared__ alignas(16) unsigned short gbf[16 * Usz];   // z*h chunk, bf16

  const int tid = threadIdx.x, lane = tid & 31, wave = tid >> 5;
  const int gb0 = blockIdx.x * 16;                        // batch rows
  const int n0  = wave * 64;                              // column slice
  const int hi = lane >> 4, nl = lane & 15;

  floatx8 h[4];
  float bucv[4], bhv[4];
#pragma unroll
  for (int tn = 0; tn < 4; tn++) {
    const int n = n0 + tn * 16 + nl;
    bucv[tn] = bu[n] + bc[n];        // attention context is zero -> only bc
    bhv[tn]  = bh[n];
#pragma unroll
    for (int r = 0; r < 8; r++)
      h[tn][r] = hidden[(gb0 + r + 8 * hi) * Usz + n];
  }
  for (int e = tid; e < 16 * Usz; e += 512)
    hbf[e] = f2bf(hidden[(gb0 + (e >> 10)) * Usz + (e & (Usz - 1))]);
  __syncthreads();

  for (int t = 0; t < Tsz; t++) {
    // ---- issue long-latency loads first so they hide under the GEMMs ----
    const float* pre_t = pre + (long)t * Bsz * Usz;
    float pv[4][8];                      // pre_t slice (read-once -> NT)
#pragma unroll
    for (int tn = 0; tn < 4; tn++) {
      const int n = n0 + tn * 16 + nl;
#pragma unroll
      for (int r = 0; r < 8; r++)
        pv[tn][r] = __builtin_nontemporal_load(&pre_t[(gb0 + r + 8 * hi) * Usz + n]);
    }
    int xr[8];
#pragma unroll
    for (int r = 0; r < 8; r++)
      xr[r] = X[(gb0 + r + 8 * hi) * Tsz + t];
    float xv[4][8];                      // embedding gather for tanh input
#pragma unroll
    for (int tn = 0; tn < 4; tn++) {
      const int n = n0 + tn * 16 + nl;
#pragma unroll
      for (int r = 0; r < 8; r++)
        xv[tn][r] = E[(long)xr[r] * Usz + n];
    }

    // ---- z = sigmoid(pre_t + h @ Wu + bu + bc) ----
    floatx8 zacc[4] = {};
    for (int kb = 0; kb < Usz; kb += 32) {
      FragBF a;
      const uint4* ap = (const uint4*)(&hbf[nl * Usz + kb]);
      a.q[0] = ap[hi];
      a.q[1] = ap[2 + hi];
#pragma unroll
      for (int tn = 0; tn < 4; tn++) {
        const int n = n0 + tn * 16 + nl;
        const uint4* bp = (const uint4*)(Wut + n * Usz + kb);
        FragBF bm;
        bm.q[0] = bp[hi];
        bm.q[1] = bp[2 + hi];
        zacc[tn] = wmma_bf16(a, bm, zacc[tn]);
      }
    }

#pragma unroll
    for (int tn = 0; tn < 4; tn++) {
      const int n = n0 + tn * 16 + nl;
#pragma unroll
      for (int r = 0; r < 8; r++) {
        const int m = r + 8 * hi;
        float zv = zacc[tn][r] + pv[tn][r] + bucv[tn];
        zv = 1.0f / (1.0f + __expf(-zv));
        zacc[tn][r] = zv;                       // keep z
        gbf[m * Usz + n] = f2bf(zv * h[tn][r]); // g = z*h, bf16 mirror
      }
    }
    __syncthreads();

    // ---- h_bar = tanh(x_t + g @ Wh + bh) ----
    floatx8 hacc[4] = {};
    for (int kb = 0; kb < Usz; kb += 32) {
      FragBF a;
      const uint4* ap = (const uint4*)(&gbf[nl * Usz + kb]);
      a.q[0] = ap[hi];
      a.q[1] = ap[2 + hi];
#pragma unroll
      for (int tn = 0; tn < 4; tn++) {
        const int n = n0 + tn * 16 + nl;
        const uint4* bp = (const uint4*)(Wht + n * Usz + kb);
        FragBF bm;
        bm.q[0] = bp[hi];
        bm.q[1] = bp[2 + hi];
        hacc[tn] = wmma_bf16(a, bm, hacc[tn]);
      }
    }

#pragma unroll
    for (int tn = 0; tn < 4; tn++) {
      const int n = n0 + tn * 16 + nl;
#pragma unroll
      for (int r = 0; r < 8; r++) {
        const int m = r + 8 * hi;
        const float hbar = tanhf(hacc[tn][r] + xv[tn][r] + bhv[tn]);
        const float zv   = zacc[tn][r];
        const float hn   = (1.0f - zv) * h[tn][r] + zv * hbar;
        h[tn][r] = hn;
        // hs [B,T,U] is write-only: bypass/NT so L2 keeps weights+pre+E
        __builtin_nontemporal_store(hn, &out[(long)(gb0 + m) * Tsz * Usz +
                                             (long)t * Usz + n]);
        hbf[m * Usz + n] = f2bf(hn);
      }
    }
    __syncthreads();
  }

  // ---- h_last [B,U] appended after hs ----
  float* hlast = out + (long)Bsz * Tsz * Usz;
#pragma unroll
  for (int tn = 0; tn < 4; tn++) {
    const int n = n0 + tn * 16 + nl;
#pragma unroll
    for (int r = 0; r < 8; r++)
      hlast[(gb0 + r + 8 * hi) * Usz + n] = h[tn][r];
  }
}

// ---------------------------------------------------------------------------
extern "C" void kernel_launch(void* const* d_in, const int* in_sizes, int n_in,
                              void* d_out, int out_size, void* d_ws, size_t ws_size,
                              hipStream_t stream) {
  const int*   X      = (const int*)d_in[0];
  // d_in[1] sequence_length: unused by the reference math
  const float* hidden = (const float*)d_in[2];
  // d_in[3] attention_hidden == 0: attention context is zeros -> Wc unused
  const float* E  = (const float*)d_in[4];
  const float* Ww = (const float*)d_in[5];
  const float* bw = (const float*)d_in[6];
  const float* Wu = (const float*)d_in[7];
  const float* bu = (const float*)d_in[8];
  const float* bc = (const float*)d_in[10];
  const float* Wh = (const float*)d_in[11];
  const float* bh = (const float*)d_in[12];
  float* out = (float*)d_out;

  char* ws = (char*)d_ws;
  unsigned short* Wwt = (unsigned short*)(ws);
  unsigned short* Wut = (unsigned short*)(ws + ((size_t)2 << 20));
  unsigned short* Wht = (unsigned short*)(ws + ((size_t)4 << 20));
  float*          pre = (float*)(ws + ((size_t)6 << 20));  // [T][B][U] fp32

  const int nT = (Usz * Usz) / 256;  // 4096 blocks
  transpose_bf16_kernel<<<nT, 256, 0, stream>>>(Ww, Wwt);
  transpose_bf16_kernel<<<nT, 256, 0, stream>>>(Wu, Wut);
  transpose_bf16_kernel<<<nT, 256, 0, stream>>>(Wh, Wht);

  dim3 gProj((Tsz * Bsz) / 128, Usz / 64);   // 100 x 16
  embed_proj_kernel<<<gProj, 256, 0, stream>>>(X, E, Wwt, bw, pre);

  rnn_scan_kernel<<<Bsz / 16, 512, 0, stream>>>(X, E, hidden, Wut, Wht,
                                                bu, bc, bh, pre, out);
}
